// ModulatedConv2d_69509750719043
// MI455X (gfx1250) — matrix operand order repaired and checked by reading, not compiled, not run on hardware
//
#include <hip/hip_runtime.h>
#include <math.h>

// ---- problem constants ----
#define B_      16
#define CIN_    256
#define COUT_   256
#define STYLE_  512
#define H_      64
#define W_      64
#define S_      4096               // H*W
#define CONV_SCALE 0.020833333333333332f   // 1/sqrt(256*9)
#define MOD_SCALE  0.04419417382415922f    // 1/sqrt(512)

typedef __attribute__((ext_vector_type(16))) _Float16 v16h;
typedef __attribute__((ext_vector_type(8)))  float    v8f;

// ---------------------------------------------------------------------------
// Kernel 1: s[b][ci] = sum_k style[b,k] * mod_w[ci,k] * MOD_SCALE + mod_b[ci]
// ---------------------------------------------------------------------------
__global__ __launch_bounds__(256) void k_style(const float* __restrict__ style,
                                               const float* __restrict__ mod_w,
                                               const float* __restrict__ mod_b,
                                               float* __restrict__ sbuf) {
  const int b  = blockIdx.x;
  const int ci = threadIdx.x;
  const float4* st = (const float4*)(style + (size_t)b * STYLE_);
  const float4* mw = (const float4*)(mod_w + (size_t)ci * STYLE_);
  float acc = 0.0f;
  #pragma unroll 4
  for (int i = 0; i < STYLE_ / 4; ++i) {
    float4 a = st[i], w = mw[i];
    acc += a.x * w.x + a.y * w.y + a.z * w.z + a.w * w.w;
  }
  sbuf[b * CIN_ + ci] = acc * MOD_SCALE + mod_b[ci];
}

// ---------------------------------------------------------------------------
// Kernel 2: per (b,co): w = CONV_SCALE*weight*s ; demod = rsqrt(sum w^2 + eps)
// write f16 weights laid out [b][tap][co][ci]   (A-matrix rows contiguous in ci)
// ---------------------------------------------------------------------------
__global__ __launch_bounds__(256) void k_modw(const float* __restrict__ weight,
                                              const float* __restrict__ sbuf,
                                              unsigned short* __restrict__ wf16) {
  const int co = blockIdx.x;
  const int b  = blockIdx.y;
  const int ci = threadIdx.x;

  const float sm = sbuf[b * CIN_ + ci] * CONV_SCALE;
  const float* wp = weight + ((size_t)co * CIN_ + ci) * 9;

  float w9[9];
  float ss = 0.0f;
  #pragma unroll
  for (int t = 0; t < 9; ++t) {
    float v = wp[t] * sm;
    w9[t] = v;
    ss += v * v;
  }

  __shared__ float red[256];
  red[ci] = ss;
  __syncthreads();
  #pragma unroll
  for (int sft = 128; sft > 0; sft >>= 1) {
    if (ci < sft) red[ci] += red[ci + sft];
    __syncthreads();
  }
  const float demod = rsqrtf(red[0] + 1e-8f);

  #pragma unroll
  for (int t = 0; t < 9; ++t) {
    union { _Float16 h; unsigned short u; } cv;
    cv.h = (_Float16)(w9[t] * demod);
    wf16[(((size_t)b * 9 + t) * COUT_ + co) * CIN_ + ci] = cv.u;
  }
}

// ---------------------------------------------------------------------------
// Kernel 3: implicit-GEMM conv.  Per block: 128 co x 128 pixels (2 rows).
// Loop: 8 ci-blocks of 32 (WMMA K)  x  3 kh-groups  x  3 dw taps.
// A is staged 3 taps at a time, so the inner dw-loop runs 24 WMMAs with no
// barrier.  x tile is staged transposed (ci fastest) once per ci-block and
// reused by all 9 taps; halo handled by an 80-B zero slot redirect.
//
// LDS layout (bytes):
//   [0     , 20560)  x tile: 257 pixel slots * 80 B (pixel = r*64+w, ci fastest,
//                    slot 256 @ 20480 is the zero slot)
//   [20560 , 51280)  A tiles: 3 taps * 128 rows * 80 B
// ---------------------------------------------------------------------------
#define XT_BASE   0
#define ZERO_OFF  20480
#define A_BASE    20560
#define A_TAP     10240

__global__ __launch_bounds__(256) void k_conv(const float* __restrict__ x,
                                              const unsigned short* __restrict__ wf16,
                                              float* __restrict__ out) {
  __shared__ __align__(16) unsigned char smem[51280];

  const int t   = threadIdx.x;
  const int n0  = blockIdx.x * 128;     // spatial tile base
  const int co0 = blockIdx.y * 128;     // cout tile base
  const int b   = blockIdx.z;
  const int h0  = n0 >> 6;              // first image row of tile

  // zero-fill the padding slot once (80 B = 20 dwords)
  if (t < 20) ((unsigned int*)(smem + ZERO_OFF))[t] = 0u;

  const int l      = t & 31;
  const int wid    = t >> 5;
  const int wave_m = wid & 3;           // 4 waves along co  (32 rows each)
  const int wave_n = wid >> 2;          // 2 waves along n   (64 cols each)
  const int l15    = l & 15;
  const int g      = l >> 4;

  v8f acc[2][4];
  #pragma unroll
  for (int i = 0; i < 2; ++i)
    #pragma unroll
    for (int j = 0; j < 4; ++j)
      #pragma unroll
      for (int k = 0; k < 8; ++k) acc[i][j][k] = 0.0f;

  // x-tile staging role: thread t handles one pixel, all 32 ci of the block.
  // Address is clamped to a valid row so loads are UNconditional; out-of-image
  // rows are zeroed with a 0/1 multiplier after the loads complete.
  const int   sp = t;                   // pixel index 0..255 (r*64 + w)
  const int   sr = sp >> 6;
  const int   sw = sp & 63;
  const int   hg = h0 - 1 + sr;
  const bool  hvb = (hg >= 0) && (hg < H_);
  const float hm  = hvb ? 1.0f : 0.0f;
  const float* xpix = x + (((size_t)b * CIN_) * H_ + (size_t)(hvb ? hg : 0)) * W_ + sw;

  // A staging role: thread t handles half a row (32 B of f16)
  const int arow  = t >> 1;
  const int ahalf = t & 1;
  const unsigned char* abase = (const unsigned char*)wf16 +
      (((size_t)b * 9 * COUT_ + (co0 + arow)) * CIN_) * 2 + ahalf * 32;

  // B-fragment geometry that does not depend on the tap
  int nl_arr[4], rb_arr[4], w0_arr[4];
  #pragma unroll
  for (int j = 0; j < 4; ++j) {
    nl_arr[j] = wave_n * 64 + j * 16 + l15;
    rb_arr[j] = nl_arr[j] >> 6;
    w0_arr[j] = nl_arr[j] & 63;
  }

  #pragma unroll 1
  for (int kb = 0; kb < CIN_ / 32; ++kb) {
    #pragma unroll 1
    for (int khg = 0; khg < 3; ++khg) {         // dh = khg - 1
      __syncthreads();   // previous group's fragment loads complete

      if (khg == 0) {
        // stage transposed x tile: [4 rows][64 w][32 ci] f16, ci contiguous.
        // 32 unconditional, coalesced loads -> one wait -> pack -> 4 b128 stores.
        const float* xs = xpix + (size_t)(kb * 32) * (H_ * W_);
        float vals[32];
        #pragma unroll
        for (int j = 0; j < 32; ++j) vals[j] = xs[(size_t)j * (H_ * W_)];
        #pragma unroll
        for (int c8 = 0; c8 < 4; ++c8) {
          union { _Float16 h[8]; uint4 u; } pk;
          #pragma unroll
          for (int j = 0; j < 8; ++j) pk.h[j] = (_Float16)(vals[c8 * 8 + j] * hm);
          *(uint4*)(smem + XT_BASE + sp * 80 + c8 * 16) = pk.u;
        }
        if (kb < CIN_ / 32 - 1)
          __builtin_prefetch(xs + (size_t)32 * (H_ * W_), 0, 1);
      }

      // stage A tiles for taps khg*3 .. khg*3+2 : [3][128 co][32 ci] f16
      #pragma unroll
      for (int dwg = 0; dwg < 3; ++dwg) {
        const int tap = khg * 3 + dwg;
        const unsigned char* asrc =
            abase + ((size_t)tap * COUT_ * CIN_ + kb * 32) * 2;
        uint4 a0 = *(const uint4*)(asrc);
        uint4 a1 = *(const uint4*)(asrc + 16);
        *(uint4*)(smem + A_BASE + dwg * A_TAP + arow * 80 + ahalf * 32)      = a0;
        *(uint4*)(smem + A_BASE + dwg * A_TAP + arow * 80 + ahalf * 32 + 16) = a1;
      }
      __syncthreads();

      // kh-dependent B geometry (uniform across the dw loop)
      int rowpix[4];
      bool hok[4];
      #pragma unroll
      for (int j = 0; j < 4; ++j) {
        const int hgl = h0 + rb_arr[j] + khg - 1;
        hok[j]    = (hgl >= 0) && (hgl < H_);
        rowpix[j] = (rb_arr[j] + khg) * 64;     // local row r = rb + dh + 1
      }

      #pragma unroll
      for (int dwg = 0; dwg < 3; ++dwg) {       // dw = dwg - 1
        // --- A fragments (16x32 f16, ISA lane layout) ---
        v16h af[2];
        #pragma unroll
        for (int i = 0; i < 2; ++i) {
          const int row = wave_m * 32 + i * 16 + l15;
          const int off = A_BASE + dwg * A_TAP + row * 80 + g * 16;
          union { uint4 u[2]; v16h h; } f;
          f.u[0] = *(const uint4*)(smem + off);        // K 8g .. 8g+7
          f.u[1] = *(const uint4*)(smem + off + 32);   // K 16+8g .. 23+8g
          af[i] = f.h;
        }

        // --- B fragments (32x16 f16): im2col gather from x tile ---
        v16h bf[4];
        #pragma unroll
        for (int j = 0; j < 4; ++j) {
          const int  wc   = w0_arr[j] + dwg - 1;
          const bool ok   = hok[j] && (wc >= 0) && (wc < W_);
          const int  base = ok ? (XT_BASE + (rowpix[j] + wc) * 80) : ZERO_OFF;
          union { uint4 u[2]; v16h h; } f;
          f.u[0] = *(const uint4*)(smem + base + g * 16);       // K 8g..8g+7
          f.u[1] = *(const uint4*)(smem + base + g * 16 + 32);  // K 16+8g..23+8g
          bf[j] = f.h;
        }

        // --- 2x4 WMMA, f32 accumulate ---
        #pragma unroll
        for (int i = 0; i < 2; ++i)
          #pragma unroll
          for (int j = 0; j < 4; ++j)
            acc[i][j] = __builtin_amdgcn_wmma_f32_16x16x32_f16(
                false, af[i], false, bf[j], (short)0, acc[i][j], false, false);
      }
    }
  }

  // epilogue: C/D layout — lane 0-15: M=v,N=lane ; lane 16-31: M=8+v,N=lane-16
  #pragma unroll
  for (int i = 0; i < 2; ++i) {
    #pragma unroll
    for (int j = 0; j < 4; ++j) {
      const int co = co0 + wave_m * 32 + i * 16 + g * 8;
      const int n  = n0 + wave_n * 64 + j * 16 + l15;
      float* op = out + ((size_t)b * COUT_ + co) * S_ + n;
      #pragma unroll
      for (int v = 0; v < 8; ++v) op[(size_t)v * S_] = acc[i][j][v];
    }
  }
}

// ---------------------------------------------------------------------------
extern "C" void kernel_launch(void* const* d_in, const int* in_sizes, int n_in,
                              void* d_out, int out_size, void* d_ws, size_t ws_size,
                              hipStream_t stream) {
  const float* x      = (const float*)d_in[0];  // [16,256,64,64]
  const float* style  = (const float*)d_in[1];  // [16,512]
  const float* weight = (const float*)d_in[2];  // [1,256,256,3,3]
  const float* mod_w  = (const float*)d_in[3];  // [256,512]
  const float* mod_b  = (const float*)d_in[4];  // [256]
  float* out = (float*)d_out;

  float*          sbuf = (float*)d_ws;                                   // 16 KB
  unsigned short* wf16 = (unsigned short*)((char*)d_ws + 16384);         // 18.9 MB

  k_style<<<B_, 256, 0, stream>>>(style, mod_w, mod_b, sbuf);
  k_modw <<<dim3(COUT_, B_), 256, 0, stream>>>(weight, sbuf, wf16);
  k_conv <<<dim3(S_ / 128, COUT_ / 128, B_), 256, 0, stream>>>(x, wf16, out);
}
